// SmallOps_47090021433859
// MI455X (gfx1250) — 8-bit matrix operand order repaired and checked by reading, compiled, not run on hardware
//
#include <hip/hip_runtime.h>
#include <hip/hip_bf16.h>


typedef int v8i __attribute__((ext_vector_type(8)));

#define NE 16
#define NT 512
#define NK 8
#define NH 2048
#define NI 1024
#define NROWS 4096
#define NROWS_PAD 4352
#define MAX_TILES 272

__global__ void moe_quantx_kernel(const float* __restrict__ x,
                                  unsigned* __restrict__ xq,
                                  float* __restrict__ s0) {
  __shared__ float red[256];
  int t = blockIdx.x, tid = threadIdx.x;
  const float* xr = x + (size_t)t * NH;
  float m = 0.f;
  for (int i = tid; i < NH; i += 256) m = fmaxf(m, fabsf(xr[i]));
  red[tid] = m; __syncthreads();
  for (int s = 128; s > 0; s >>= 1) {
    if (tid < s) red[tid] = fmaxf(red[tid], red[tid + s]);
    __syncthreads();
  }
  float mx = red[0];
  float inv = mx > 0.f ? 127.f / mx : 0.f;
  if (tid == 0) s0[t] = mx / 127.f;
  for (int d = tid; d < NH / 4; d += 256) {
    unsigned w = 0;
    #pragma unroll
    for (int b = 0; b < 4; b++) {
      int q = (int)rintf(xr[d * 4 + b] * inv);
      q = q < -128 ? -128 : (q > 127 ? 127 : q);
      w |= ((unsigned)(q & 0xFF)) << (8 * b);
    }
    xq[(size_t)t * (NH / 4) + d] = w;
  }
}

__global__ void moe_bucket_kernel(const int* __restrict__ eids,
                                  int* __restrict__ srow,
                                  int* __restrict__ tExp,
                                  int* __restrict__ tBase) {
  if (threadIdx.x != 0 || blockIdx.x != 0) return;
  int counts[NE], start[NE], cur[NE];
  for (int e = 0; e < NE; e++) counts[e] = 0;
  for (int i = 0; i < NROWS; i++) counts[eids[i]]++;
  int base = 0;
  for (int e = 0; e < NE; e++) { start[e] = base; base += ((counts[e] + 15) >> 4) << 4; }
  for (int p = 0; p < NROWS_PAD; p++) srow[p] = -1;
  int tile = 0;
  for (int e = 0; e < NE; e++) {
    int padded = ((counts[e] + 15) >> 4) << 4;
    for (int j = 0; j < padded; j += 16) { tExp[tile] = e; tBase[tile] = start[e] + j; tile++; }
  }
  for (; tile < MAX_TILES; tile++) { tExp[tile] = -1; tBase[tile] = 0; }
  for (int e = 0; e < NE; e++) cur[e] = start[e];
  for (int i = 0; i < NROWS; i++) { int e = eids[i]; srow[cur[e]++] = i; }
}

__global__ void moe_packw_kernel(const int* __restrict__ src, unsigned* __restrict__ dst,
                                 int KT, int NTl, int Kdim, int N) {
  long id = (long)blockIdx.x * 256 + threadIdx.x;
  int v = (int)(id & 7);
  int lane = (int)((id >> 3) & 31);
  long rem = id >> 8;
  int kt = (int)(rem % KT); rem /= KT;
  int nt = (int)(rem % NTl);
  int e = (int)(rem / NTl);
  int n = nt * 16 + (lane & 15);
  int kb = kt * 64 + (v & 3) * 4 + ((v >> 2) & 1) * 32 + ((lane >> 4) & 1) * 16;
  unsigned w = 0;
  #pragma unroll
  for (int b = 0; b < 4; b++) {
    int val = src[((long)e * Kdim + (kb + b)) * (long)N + n];
    w |= ((unsigned)(val & 0xFF)) << (8 * b);
  }
  dst[id] = w;
}

__device__ __forceinline__ v8i load_afrag(const unsigned* Alds, int rowStride,
                                          int c, int half, int kt) {
  v8i a;
  #pragma unroll
  for (int q = 0; q < 4; q++) {
    uint2 p = *(const uint2*)&Alds[c * rowStride + kt * 16 + half * 4 + (q & 1) * 2 + (q >> 1) * 8];
    a[2 * q] = (int)p.x; a[2 * q + 1] = (int)p.y;
  }
  return a;
}

__device__ __forceinline__ v8i load_bfrag(const unsigned* bp) {
  const uint4* p = (const uint4*)bp;
  uint4 b0 = p[0], b1 = p[1];
  v8i b;
  b[0] = (int)b0.x; b[1] = (int)b0.y; b[2] = (int)b0.z; b[3] = (int)b0.w;
  b[4] = (int)b1.x; b[5] = (int)b1.y; b[6] = (int)b1.z; b[7] = (int)b1.w;
  return b;
}

__global__ void __launch_bounds__(256)
moe_gemm1_kernel(const unsigned* __restrict__ xq, const float* __restrict__ s0,
                 const int* __restrict__ srow, const int* __restrict__ tExp,
                 const int* __restrict__ tBase,
                 const unsigned* __restrict__ w1p, const float* __restrict__ ws1,
                 unsigned* __restrict__ aq, float* __restrict__ s1buf) {
  __shared__ unsigned Alds[16 * 512];
  __shared__ unsigned actq[16 * 256];
  __shared__ int rowids[16];
  __shared__ float s0row[16];
  __shared__ unsigned rowmaxU[16];
  __shared__ float invrow[16];

  int tile = blockIdx.x;
  int e = tExp[tile];
  if (e < 0) return;
  int base = tBase[tile];
  int tid = threadIdx.x;
  if (tid < 16) { rowids[tid] = srow[base + tid]; rowmaxU[tid] = 0u; }
  __syncthreads();
  for (int idx = tid; idx < 16 * 512; idx += 256) {
    int row = idx >> 9, d = idx & 511;
    int r = rowids[row];
    Alds[idx] = (r >= 0) ? xq[(size_t)(r >> 3) * 512 + d] : 0u;
  }
  if (tid < 16) s0row[tid] = (rowids[tid] >= 0) ? s0[rowids[tid] >> 3] : 0.f;
  __syncthreads();

  int wave = tid >> 5, lane = tid & 31, half = lane >> 4, c = lane & 15;
  float s0v[8];
  #pragma unroll
  for (int j = 0; j < 8; j++) s0v[j] = s0row[j + 8 * half];

  float actv[8][8];
  float rmax[8];
  #pragma unroll
  for (int j = 0; j < 8; j++) rmax[j] = 0.f;

  #pragma unroll
  for (int i = 0; i < 8; i++) {
    int nt = wave * 8 + i;
    v8i accL, accR;
    #pragma unroll
    for (int j = 0; j < 8; j++) { accL[j] = 0; accR[j] = 0; }
    const unsigned* bpl = w1p + ((size_t)(e * 128 + nt) * 32) * 256 + lane * 8;
    const unsigned* bpr = w1p + ((size_t)(e * 128 + 64 + nt) * 32) * 256 + lane * 8;
    v8i bl0 = load_bfrag(bpl);
    v8i br0 = load_bfrag(bpr);
    v8i bl1 = load_bfrag(bpl + 256);
    v8i br1 = load_bfrag(bpr + 256);
    #pragma unroll 1
    for (int kt = 0; kt < 32; kt += 2) {
      v8i a0 = load_afrag(Alds, 512, c, half, kt);
      accL = __builtin_amdgcn_wmma_i32_16x16x64_iu8(true, a0, true, bl0, accL, false, false);
      accR = __builtin_amdgcn_wmma_i32_16x16x64_iu8(true, a0, true, br0, accR, false, false);
      __builtin_amdgcn_sched_barrier(0);
      bl0 = load_bfrag(bpl + 512);
      br0 = load_bfrag(bpr + 512);
      __builtin_prefetch(bpl + 1024, 0, 1);
      v8i a1 = load_afrag(Alds, 512, c, half, kt + 1);
      accL = __builtin_amdgcn_wmma_i32_16x16x64_iu8(true, a1, true, bl1, accL, false, false);
      accR = __builtin_amdgcn_wmma_i32_16x16x64_iu8(true, a1, true, br1, accR, false, false);
      __builtin_amdgcn_sched_barrier(0);
      bl1 = load_bfrag(bpl + 768);
      br1 = load_bfrag(bpr + 768);
      __builtin_prefetch(bpr + 1024, 0, 1);
      bpl += 512; bpr += 512;
    }
    float wsl = ws1[e * (2 * NI) + nt * 16 + c];
    float wsr = ws1[e * (2 * NI) + NI + nt * 16 + c];
    #pragma unroll
    for (int j = 0; j < 8; j++) {
      float lf = (float)accL[j] * wsl * s0v[j];
      float rf = (float)accR[j] * wsr * s0v[j];
      float sig = 1.f / (1.f + __expf(-lf));
      float av = lf * sig * rf;
      actv[i][j] = av;
      rmax[j] = fmaxf(rmax[j], fabsf(av));
    }
  }
  #pragma unroll
  for (int j = 0; j < 8; j++)
    atomicMax(&rowmaxU[j + 8 * half], __float_as_uint(rmax[j]));
  __syncthreads();
  if (tid < 16) {
    float m = __uint_as_float(rowmaxU[tid]);
    s1buf[base + tid] = m / 127.f;
    invrow[tid] = (m > 0.f) ? 127.f / m : 0.f;
  }
  __syncthreads();
  float invv[8];
  #pragma unroll
  for (int j = 0; j < 8; j++) invv[j] = invrow[j + 8 * half];
  char* actb = (char*)actq;
  #pragma unroll
  for (int i = 0; i < 8; i++) {
    int nt = wave * 8 + i;
    #pragma unroll
    for (int j = 0; j < 8; j++) {
      int q = (int)rintf(actv[i][j] * invv[j]);
      q = q < -128 ? -128 : (q > 127 ? 127 : q);
      actb[(j + 8 * half) * NI + nt * 16 + c] = (char)q;
    }
  }
  __syncthreads();
  for (int idx = tid; idx < 16 * 256; idx += 256)
    aq[(size_t)(base + (idx >> 8)) * 256 + (idx & 255)] = actq[idx];
}

__global__ void __launch_bounds__(256)
moe_gemm2_kernel(const unsigned* __restrict__ aq, const float* __restrict__ s1buf,
                 const int* __restrict__ srow, const int* __restrict__ tExp,
                 const int* __restrict__ tBase,
                 const unsigned* __restrict__ w2p, const float* __restrict__ ws2,
                 float* __restrict__ y2) {
  __shared__ unsigned Alds[16 * 256];
  __shared__ int rowids[16];
  __shared__ float s1row[16];
  int tile = blockIdx.x;
  int e = tExp[tile];
  if (e < 0) return;
  int base = tBase[tile];
  int tid = threadIdx.x;
  if (tid < 16) { rowids[tid] = srow[base + tid]; s1row[tid] = s1buf[base + tid]; }
  for (int idx = tid; idx < 16 * 256; idx += 256)
    Alds[idx] = aq[(size_t)base * 256 + idx];
  __syncthreads();

  int wave = tid >> 5, lane = tid & 31, half = lane >> 4, c = lane & 15;
  float s1v[8]; int rid[8];
  #pragma unroll
  for (int j = 0; j < 8; j++) { s1v[j] = s1row[j + 8 * half]; rid[j] = rowids[j + 8 * half]; }

  #pragma unroll 1
  for (int i = 0; i < 16; i++) {
    int nt = wave * 16 + i;
    v8i acc;
    #pragma unroll
    for (int j = 0; j < 8; j++) acc[j] = 0;
    const unsigned* bb = w2p + ((size_t)(e * 128 + nt) * 16) * 256 + lane * 8;
    v8i b0 = load_bfrag(bb);
    v8i b1 = load_bfrag(bb + 256);
    #pragma unroll 1
    for (int kt = 0; kt < 16; kt += 2) {
      v8i a0 = load_afrag(Alds, 256, c, half, kt);
      acc = __builtin_amdgcn_wmma_i32_16x16x64_iu8(true, a0, true, b0, acc, false, false);
      __builtin_amdgcn_sched_barrier(0);
      b0 = load_bfrag(bb + 512);
      __builtin_prefetch(bb + 1024, 0, 1);
      v8i a1 = load_afrag(Alds, 256, c, half, kt + 1);
      acc = __builtin_amdgcn_wmma_i32_16x16x64_iu8(true, a1, true, b1, acc, false, false);
      __builtin_amdgcn_sched_barrier(0);
      b1 = load_bfrag(bb + 768);
      bb += 512;
    }
    float wc = ws2[e * NH + nt * 16 + c];
    #pragma unroll
    for (int j = 0; j < 8; j++) {
      int r = rid[j];
      if (r >= 0) y2[(size_t)r * NH + nt * 16 + c] = (float)acc[j] * wc * s1v[j];
    }
  }
}

__global__ void moe_combine_kernel(const float* __restrict__ y2,
                                   const float* __restrict__ es,
                                   float* __restrict__ out) {
  int idx = blockIdx.x * 256 + threadIdx.x;
  int t = idx >> 11, h = idx & (NH - 1);
  float acc = 0.f;
  #pragma unroll
  for (int k = 0; k < NK; k++)
    acc += es[t * NK + k] * y2[(size_t)(t * NK + k) * NH + h];
  out[idx] = acc;
}

extern "C" void kernel_launch(void* const* d_in, const int* in_sizes, int n_in,
                              void* d_out, int out_size, void* d_ws, size_t ws_size,
                              hipStream_t stream) {
  const float* x   = (const float*)d_in[0];
  const int* eids  = (const int*)d_in[1];
  const int* w1    = (const int*)d_in[2];
  const float* ws1 = (const float*)d_in[3];
  const int* w2    = (const int*)d_in[4];
  const float* ws2 = (const float*)d_in[5];
  const float* es  = (const float*)d_in[7];
  float* out = (float*)d_out;

  size_t off = 0;
  auto carve = [&](size_t bytes) -> void* {
    void* p = (char*)d_ws + off;
    off += (bytes + 255) & ~(size_t)255;
    return p;
  };
  unsigned* xq   = (unsigned*)carve((size_t)NT * (NH / 4) * 4);
  float*    s0   = (float*)   carve((size_t)NT * 4);
  int*      srow = (int*)     carve((size_t)NROWS_PAD * 4);
  int*      tExp = (int*)     carve((size_t)MAX_TILES * 4);
  int*      tBas = (int*)     carve((size_t)MAX_TILES * 4);
  unsigned* w1p  = (unsigned*)carve((size_t)NE * 128 * 32 * 256 * 4);
  unsigned* w2p  = (unsigned*)carve((size_t)NE * 128 * 16 * 256 * 4);
  unsigned* aqb  = (unsigned*)carve((size_t)NROWS_PAD * 256 * 4);
  float*    s1b  = (float*)   carve((size_t)NROWS_PAD * 4);
  float*    y2   = (float*)   carve((size_t)NROWS * NH * 4);

  moe_packw_kernel<<<65536, 256, 0, stream>>>(w1, w1p, 32, 128, NH, 2 * NI);
  moe_packw_kernel<<<32768, 256, 0, stream>>>(w2, w2p, 16, 128, NI, NH);
  moe_quantx_kernel<<<NT, 256, 0, stream>>>(x, xq, s0);
  moe_bucket_kernel<<<1, 64, 0, stream>>>(eids, srow, tExp, tBas);
  moe_gemm1_kernel<<<MAX_TILES, 256, 0, stream>>>(xq, s0, srow, tExp, tBas, w1p, ws1, aqb, s1b);
  moe_gemm2_kernel<<<MAX_TILES, 256, 0, stream>>>(aqb, s1b, srow, tExp, tBas, w2p, ws2, y2);
  moe_combine_kernel<<<(NT * NH) / 256, 256, 0, stream>>>(y2, es, out);
}
